// GloLoConv_5866925326454
// MI455X (gfx1250) — compile-verified, matrix operand order checked by reading, not run on hardware
//
#include <hip/hip_runtime.h>

typedef __attribute__((ext_vector_type(16))) _Float16 v16h;
typedef __attribute__((ext_vector_type(8)))  _Float16 v8h;
typedef __attribute__((ext_vector_type(8)))  float    v8f;

#define NODES_TOTAL 16384   // N
#define IN_CH 128
#define HC 256              // HEADS*OUT_C
#define NE 262144           // E
#define NROWS (NE + NODES_TOTAL)
#define LIN_K 16384
#define LIN_N 4096
#define KSPLIT 8
#define OUT_ELEMS (256 * LIN_N)

// ---------- fp32 -> f16 convert ----------
__global__ void k_cvt_f16(const float* __restrict__ s, _Float16* __restrict__ d, int n) {
    int i = blockIdx.x * blockDim.x + threadIdx.x;
    if (i < n) d[i] = (_Float16)s[i];
}

// ---------- small transpose+convert: s[rows,cols] -> d[cols,rows] f16 ----------
__global__ void k_tr_small(const float* __restrict__ s, _Float16* __restrict__ d,
                           int rows, int cols) {
    int i = blockIdx.x * blockDim.x + threadIdx.x;
    if (i < rows * cols) {
        int r = i / cols, c = i % cols;
        d[c * rows + r] = (_Float16)s[i];
    }
}

// ---------- big transpose+convert: lin_W [LIN_K, LIN_N] f32 -> Wt [LIN_N, LIN_K] f16 ----------
__global__ void k_tr_lin(const float* __restrict__ W, _Float16* __restrict__ Wt) {
    __shared__ float t[32][33];
    int k = blockIdx.y * 32 + threadIdx.y;
    int n = blockIdx.x * 32 + threadIdx.x;
    t[threadIdx.y][threadIdx.x] = W[(size_t)k * LIN_N + n];
    __syncthreads();
    int n2 = blockIdx.x * 32 + threadIdx.y;
    int k2 = blockIdx.y * 32 + threadIdx.x;
    Wt[(size_t)n2 * LIN_K + k2] = (_Float16)t[threadIdx.x][threadIdx.y];
}

// ---------- register-tiled WMMA GEMM (compile-time shapes) ----------
// Each wave computes a 64x64 output tile: 4 A-fragments x 4 B-fragments,
// 16 x v_wmma_f32_16x16x32_f16 per 32-wide k-step (1 b128 load per WMMA).
// K-split (KS>1): slice s accumulates K range [s*K/KS, (s+1)*K/KS) into
// partial buffer C + s*M*N (bias applied by the reducer instead).
// Fragment layout per CDNA5 ISA 7.12.2 (wave32): lane L holds row (L&15),
// K-half selected by (L>>4); C/D VGPR v -> row v + 8*(L>>4).
// __launch_bounds__(256,1): allow full VGPR budget (128 acc VGPRs live).
template<int M, int N, int K, int KS>
__global__ void __launch_bounds__(256, 1)
k_wmma_gemm64(const _Float16* __restrict__ A, const _Float16* __restrict__ Bt,
              const float* __restrict__ bias, float* __restrict__ C) {
    constexpr int TM = M / 64;
    constexpr int TN = N / 64;
    constexpr int TILES = TM * TN;
    constexpr int KC = K / KS;

    int wave = (blockIdx.x * blockDim.x + threadIdx.x) >> 5;
    int lane = threadIdx.x & 31;
    int slice = wave / TILES;
    if (slice >= KS) return;
    int t  = wave - slice * TILES;
    int mt = t / TN, nt = t % TN;

    int hs = lane >> 4, idx = lane & 15;
    const _Float16* ap[4];
    const _Float16* bp[4];
#pragma unroll
    for (int i = 0; i < 4; ++i) {
        ap[i] = A  + (size_t)(mt * 64 + i * 16 + idx) * K + (size_t)slice * KC + hs * 8;
        bp[i] = Bt + (size_t)(nt * 64 + i * 16 + idx) * K + (size_t)slice * KC + hs * 8;
    }

    v8f acc[4][4] = {};
    for (int k0 = 0; k0 < KC; k0 += 32) {
        // prefetch the streamed B rows ahead (-> global_prefetch_b8)
        __builtin_prefetch(bp[0] + k0 + 512, 0, 0);
        v16h af[4];
#pragma unroll
        for (int i = 0; i < 4; ++i) {
            v8h a0 = *(const v8h*)(ap[i] + k0);
            v8h a1 = *(const v8h*)(ap[i] + k0 + 16);
            af[i] = __builtin_shufflevector(a0, a1,
                     0,1,2,3,4,5,6,7,8,9,10,11,12,13,14,15);
        }
#pragma unroll
        for (int j = 0; j < 4; ++j) {
            v8h b0 = *(const v8h*)(bp[j] + k0);
            v8h b1 = *(const v8h*)(bp[j] + k0 + 16);
            v16h bf = __builtin_shufflevector(b0, b1,
                       0,1,2,3,4,5,6,7,8,9,10,11,12,13,14,15);
#pragma unroll
            for (int i = 0; i < 4; ++i)
                acc[i][j] = __builtin_amdgcn_wmma_f32_16x16x32_f16(
                    false, af[i], false, bf, (short)0, acc[i][j], false, false);
        }
    }

    if (KS == 1) {
#pragma unroll
        for (int i = 0; i < 4; ++i) {
            int r0 = mt * 64 + i * 16 + hs * 8;
#pragma unroll
            for (int j = 0; j < 4; ++j) {
                int col = nt * 64 + j * 16 + idx;
                float bb = bias ? bias[col] : 0.0f;
#pragma unroll
                for (int v = 0; v < 8; ++v)
                    C[(size_t)(r0 + v) * N + col] = acc[i][j][v] + bb;
            }
        }
    } else {
        float* Cp = C + (size_t)slice * M * N;
#pragma unroll
        for (int i = 0; i < 4; ++i) {
            int r0 = mt * 64 + i * 16 + hs * 8;
#pragma unroll
            for (int j = 0; j < 4; ++j) {
                int col = nt * 64 + j * 16 + idx;
#pragma unroll
                for (int v = 0; v < 8; ++v)
                    Cp[(size_t)(r0 + v) * N + col] = acc[i][j][v];
            }
        }
    }
}

// ---------- deterministic K-split reduction + bias ----------
__global__ void k_reduce8(const float* __restrict__ Cp, const float* __restrict__ bias,
                          float* __restrict__ out) {
    int i = blockIdx.x * blockDim.x + threadIdx.x;
    if (i >= OUT_ELEMS) return;
    float s = 0.f;
#pragma unroll
    for (int k = 0; k < KSPLIT; ++k) s += Cp[(size_t)k * OUT_ELEMS + i];
    out[i] = s + bias[i & (LIN_N - 1)];
}

// ---------- deterministic column mean of edge_attr, pass 1 ----------
__global__ void k_mean_part(const float* __restrict__ ea, float* __restrict__ part) {
    __shared__ float lds[256];
    int t = threadIdx.x;
    int k = t & 15, g = t >> 4;
    int base = blockIdx.x * 256 + g * 16;
    float acc = 0.f;
#pragma unroll
    for (int rr = 0; rr < 16; ++rr) acc += ea[(size_t)(base + rr) * 16 + k];
    lds[t] = acc;
    __syncthreads();
    if (t < 16) {
        float s = 0.f;
        for (int g2 = 0; g2 < 16; ++g2) s += lds[g2 * 16 + t];
        part[blockIdx.x * 16 + t] = s;
    }
}
// ---------- pass 2 ----------
__global__ void k_mean_fin(const float* __restrict__ part, float* __restrict__ mean) {
    int t = threadIdx.x;
    if (t < 16) {
        float s = 0.f;
        for (int b = 0; b < 1024; ++b) s += part[b * 16 + t];
        mean[t] = s / (float)NE;
    }
}

// ---------- per-edge-row attention logits (ee GEMM folded in) ----------
__global__ void k_logits(const int* __restrict__ ei, const float* __restrict__ eattr,
                         const float* __restrict__ mean_ea, const float* __restrict__ We,
                         const float* __restrict__ att, const float* __restrict__ xl,
                         const float* __restrict__ xr, float* __restrict__ logits) {
    int row  = (blockIdx.x * blockDim.x + threadIdx.x) >> 5;
    int lane = threadIdx.x & 31;
    if (row >= NROWS) return;
    int s, d;
    float ea[16];
    if (row < NE) {
        s = ei[row];
        d = ei[NE + row];
#pragma unroll
        for (int k = 0; k < 16; ++k) ea[k] = eattr[(size_t)row * 16 + k];
    } else {
        s = d = row - NE;
#pragma unroll
        for (int k = 0; k < 16; ++k) ea[k] = mean_ea[k];
    }
    int j0 = lane * 8;
    const float* xls = xl + (size_t)s * HC + j0;
    const float* xrd = xr + (size_t)d * HC + j0;
    float partial = 0.f;
#pragma unroll
    for (int jj = 0; jj < 8; ++jj) {
        int j = j0 + jj;
        float ee = 0.f;
#pragma unroll
        for (int k = 0; k < 16; ++k) ee = fmaf(ea[k], We[k * HC + j], ee);
        float f = xls[jj] + xrd[jj] + ee;
        f = f > 0.f ? f : 0.2f * f;            // leaky_relu, slope 0.2
        partial = fmaf(f, att[j], partial);
    }
    partial += __shfl_xor(partial, 1, 32);
    partial += __shfl_xor(partial, 2, 32);
    partial += __shfl_xor(partial, 4, 32);
    if ((lane & 7) == 0) logits[(size_t)row * 4 + (lane >> 3)] = partial;
}

// ---------- per-node softmax over its 17 fixed rows + weighted aggregation ----------
__global__ void k_aggr(const int* __restrict__ ei, const float* __restrict__ logits,
                       const float* __restrict__ xl, const float* __restrict__ bgat,
                       _Float16* __restrict__ h16) {
    int n    = (blockIdx.x * blockDim.x + threadIdx.x) >> 5;
    int lane = threadIdx.x & 31;
    if (n >= NODES_TOTAL) return;
    int hd = lane >> 3;
    float lg[17];
    float m = -1e30f;
#pragma unroll
    for (int i = 0; i < 16; ++i) {
        lg[i] = logits[(size_t)(n * 16 + i) * 4 + hd];
        m = fmaxf(m, lg[i]);
    }
    lg[16] = logits[(size_t)(NE + n) * 4 + hd];
    m = fmaxf(m, lg[16]);
    float den = 0.f;
#pragma unroll
    for (int i = 0; i < 17; ++i) { lg[i] = __expf(lg[i] - m); den += lg[i]; }
    float inv = 1.0f / den;
    int j0 = lane * 8;
    float acc[8];
#pragma unroll
    for (int jj = 0; jj < 8; ++jj) acc[jj] = 0.f;
#pragma unroll
    for (int i = 0; i < 17; ++i) {
        int s = (i < 16) ? ei[n * 16 + i] : n;
        float w = lg[i] * inv;
        const float* xs = xl + (size_t)s * HC + j0;
#pragma unroll
        for (int jj = 0; jj < 8; ++jj) acc[jj] = fmaf(w, xs[jj], acc[jj]);
    }
    const float invc = 1.0f / 17.0f;   // aggr='mean', uniform degree 17
#pragma unroll
    for (int jj = 0; jj < 8; ++jj)
        h16[(size_t)n * HC + j0 + jj] = (_Float16)(acc[jj] * invc + bgat[j0 + jj]);
}

extern "C" void kernel_launch(void* const* d_in, const int* in_sizes, int n_in,
                              void* d_out, int out_size, void* d_ws, size_t ws_size,
                              hipStream_t stream) {
    const float* x     = (const float*)d_in[0];
    const int*   ei    = (const int*)d_in[1];     // [2, E] : src then dst
    const float* eattr = (const float*)d_in[2];
    const float* Wl    = (const float*)d_in[4];
    const float* Wr    = (const float*)d_in[5];
    const float* We    = (const float*)d_in[6];
    const float* att   = (const float*)d_in[7];
    const float* bgat  = (const float*)d_in[8];
    const float* linW  = (const float*)d_in[9];
    const float* linb  = (const float*)d_in[10];
    float* out = (float*)d_out;

    // workspace carve-out (256B aligned)
    char* ws = (char*)d_ws;
    size_t off = 0;
    auto carve = [&](size_t bytes) -> char* {
        char* p = ws + off;
        off = (off + bytes + 255) & ~(size_t)255;
        return p;
    };
    _Float16* x16   = (_Float16*)carve((size_t)NODES_TOTAL * IN_CH * 2);
    _Float16* Wlt   = (_Float16*)carve((size_t)HC * IN_CH * 2);
    _Float16* Wrt   = (_Float16*)carve((size_t)HC * IN_CH * 2);
    float*    xl    = (float*)   carve((size_t)NODES_TOTAL * HC * 4);
    float*    xr    = (float*)   carve((size_t)NODES_TOTAL * HC * 4);
    float*    lgts  = (float*)   carve((size_t)NROWS * 4 * 4);
    float*    part  = (float*)   carve((size_t)1024 * 16 * 4);
    float*    meanE = (float*)   carve(64);
    _Float16* h16   = (_Float16*)carve((size_t)NODES_TOTAL * HC * 2);
    _Float16* Wt    = (_Float16*)carve((size_t)LIN_N * LIN_K * 2);
    float*    Cpart = (float*)   carve((size_t)KSPLIT * OUT_ELEMS * 4);

    // 1) input conversions
    {
        int n = NODES_TOTAL * IN_CH;
        k_cvt_f16<<<(n + 255) / 256, 256, 0, stream>>>(x, x16, n);
    }
    k_tr_small<<<(IN_CH * HC + 255) / 256, 256, 0, stream>>>(Wl, Wlt, IN_CH, HC);
    k_tr_small<<<(IN_CH * HC + 255) / 256, 256, 0, stream>>>(Wr, Wrt, IN_CH, HC);
    k_tr_lin<<<dim3(LIN_N / 32, LIN_K / 32), dim3(32, 32), 0, stream>>>(linW, Wt);

    // 2) xl = x @ W_l, xr = x @ W_r : 64x64 tiles, no K-split
    {
        int waves = (NODES_TOTAL / 64) * (HC / 64);           // 1024
        k_wmma_gemm64<NODES_TOTAL, HC, IN_CH, 1>
            <<<waves / 8, 256, 0, stream>>>(x16, Wlt, nullptr, xl);
        k_wmma_gemm64<NODES_TOTAL, HC, IN_CH, 1>
            <<<waves / 8, 256, 0, stream>>>(x16, Wrt, nullptr, xr);
    }

    // 3) mean edge_attr (deterministic two-pass)
    k_mean_part<<<1024, 256, 0, stream>>>(eattr, part);
    k_mean_fin<<<1, 256, 0, stream>>>(part, meanE);

    // 4) per-edge-row logits
    k_logits<<<NROWS / 8, 256, 0, stream>>>(ei, eattr, meanE, We, att, xl, xr, lgts);

    // 5) per-node softmax + aggregation -> h16 [N, HC] (== [256, 16384] flat)
    k_aggr<<<NODES_TOTAL / 8, 256, 0, stream>>>(ei, lgts, xl, bgat, h16);

    // 6) y = h @ lin_W : 64x64 tiles, K split 8 ways into partials
    {
        int waves = (256 / 64) * (LIN_N / 64) * KSPLIT;       // 2048
        k_wmma_gemm64<256, LIN_N, LIN_K, KSPLIT>
            <<<waves / 8, 256, 0, stream>>>(h16, Wt, nullptr, Cpart);
    }
    // 7) partial reduction + lin_b -> d_out ([256,4096] row-major == [16384,64])
    k_reduce8<<<OUT_ELEMS / 256, 256, 0, stream>>>(Cpart, linb, out);
}
